// BayesLSTM_37589553775217
// MI455X (gfx1250) — matrix operand order repaired and checked by reading, not compiled, not run on hardware
//
#include <hip/hip_runtime.h>
#include <hip/hip_bf16.h>
#include <math.h>

// ---------------------------------------------------------------------------
// Problem constants
// ---------------------------------------------------------------------------
#define Bsz 256
#define Ssz 512
#define Isz 64
#define Hsz 128
#define Gsz 4
#define KD  (Isz + Hsz)          // 192  reduction dim
#define ND  (Gsz * Hsz)          // 512  gate-output dim
#define WSTEP (Gsz * KD * Hsz)   // 98304 elements of W per timestep
#define BT  16                   // batch rows per workgroup
#define NWG (Bsz / BT)           // 16 workgroups

// workspace layout (bytes, all 256B aligned)
#define OFF_WALL  ((size_t)0)                       // bf16 W_all [S][G][KD][H]  100,663,296 B
#define OFF_BALL  ((size_t)100663296)               // f32  b_all [S][G][H]        1,048,576 B
#define OFF_XBF   ((size_t)101711872)               // bf16 x_bf  [S][B][I]       16,777,216 B
#define OFF_HFIN  ((size_t)118489088)               // f32  h_final [B][H]           131,072 B
#define OFF_WSIG  ((size_t)118620160)               // f32  softplus(w_rho)          393,216 B

typedef __attribute__((ext_vector_type(16))) __bf16 v16bf;
typedef __attribute__((ext_vector_type(8)))  __bf16 v8bf;
typedef __attribute__((ext_vector_type(8)))  float  v8f;

__device__ __forceinline__ unsigned short f2bf(float f) {
    union { float f; unsigned u; } v; v.f = f;
    unsigned r = v.u + 0x7FFFu + ((v.u >> 16) & 1u);   // round-to-nearest-even
    return (unsigned short)(r >> 16);
}
__device__ __forceinline__ float softplusf(float x) {
    return x > 20.f ? x : log1pf(expf(x));
}
__device__ __forceinline__ float sigmf(float x) {
    return 1.f / (1.f + expf(-x));
}

// ---------------------------------------------------------------------------
// prep 1: w_sigma = softplus(w_rho)   (98,304 elements)
// ---------------------------------------------------------------------------
__global__ __launch_bounds__(256) void k_prep_sigma(const float* __restrict__ w_rho,
                                                    float* __restrict__ w_sig) {
    int gid = blockIdx.x * 256 + threadIdx.x;       // exactly 98,304 threads
    w_sig[gid] = softplusf(w_rho[gid]);
}

// ---------------------------------------------------------------------------
// prep 2: W_all[t,g,k,h] = bf16(w_mu + w_sigma * eps_w[t])  — the 201MB stream
// 2 elements / thread, packed bf16x2 stores.
// ---------------------------------------------------------------------------
__global__ __launch_bounds__(256) void k_prep_w(const float* __restrict__ w_mu,
                                                const float* __restrict__ w_sig,
                                                const float* __restrict__ eps_w,
                                                unsigned* __restrict__ wall_u32) {
    size_t gid  = (size_t)blockIdx.x * 256 + threadIdx.x; // 25,165,824 threads
    size_t idx2 = gid * 2;
    unsigned rem = (unsigned)(idx2 % (size_t)WSTEP);
    const float2 ev = *(const float2*)(eps_w + idx2);
    const float2 mu = *(const float2*)(w_mu + rem);
    const float2 sg = *(const float2*)(w_sig + rem);
    unsigned short b0 = f2bf(fmaf(sg.x, ev.x, mu.x));
    unsigned short b1 = f2bf(fmaf(sg.y, ev.y, mu.y));
    wall_u32[gid] = (unsigned)b0 | ((unsigned)b1 << 16);
}

// ---------------------------------------------------------------------------
// prep 3: b_all[t,g,h] = b_mu + softplus(b_rho)*eps_b   (262,144 elements)
// ---------------------------------------------------------------------------
__global__ __launch_bounds__(256) void k_prep_b(const float* __restrict__ b_mu,
                                                const float* __restrict__ b_rho,
                                                const float* __restrict__ eps_b,
                                                float* __restrict__ ball) {
    int gid = blockIdx.x * 256 + threadIdx.x;       // 262,144 threads
    int gh  = gid % (Gsz * Hsz);
    ball[gid] = fmaf(softplusf(b_rho[gh]), eps_b[gid], b_mu[gh]);
}

// ---------------------------------------------------------------------------
// prep 4: x_bf[t,b,i] = bf16(x[b,t,i])  — time-major bf16 copy of x
// ---------------------------------------------------------------------------
__global__ __launch_bounds__(256) void k_prep_x(const float* __restrict__ x,
                                                unsigned* __restrict__ xbf_u32) {
    size_t gid = (size_t)blockIdx.x * 256 + threadIdx.x; // 4,194,304 threads
    size_t p   = gid * 2;                                // output element index
    int t = (int)(p / (Bsz * Isz));
    int r = (int)(p % (Bsz * Isz));
    int b = r / Isz;
    int i = r % Isz;
    const float2 v = *(const float2*)(x + ((size_t)b * Ssz + t) * Isz + i);
    xbf_u32[gid] = (unsigned)f2bf(v.x) | ((unsigned)f2bf(v.y) << 16);
}

// ---------------------------------------------------------------------------
// main recurrent kernel: 16 WGs x 256 threads (8 waves), BT=16 rows per WG.
// Per step: build bf16 [16 x 192] A-tile in LDS, GEMM vs W_all[t] (bf16,
// streamed through L2) with v_wmma_f32_16x16x32_bf16, then gate nonlinearity
// + c/h update with state resident in LDS.
// ---------------------------------------------------------------------------
#define CPAD 200   // comb row pitch (elements); 400B per row, 16B aligned
__global__ __launch_bounds__(256) void k_lstm(const unsigned short* __restrict__ xbf,
                                              const unsigned short* __restrict__ wall,
                                              const float* __restrict__ ball,
                                              const float* __restrict__ h0,
                                              const float* __restrict__ c0,
                                              float* __restrict__ hfin) {
    __shared__ unsigned short comb[BT * CPAD];   //  6,400 B  bf16 [x_t | h]
    __shared__ float hS[BT * Hsz];               //  8,192 B
    __shared__ float cS[BT * Hsz];               //  8,192 B
    __shared__ float gO[BT * ND];                // 32,768 B  pre-activation gates

    const int tid  = threadIdx.x;
    const int r0   = blockIdx.x * BT;
    const int lane = tid & 31;
    const int wave = tid >> 5;
    const int g       = wave >> 1;          // gate 0..3 per wave pair
    const int colHalf = (wave & 1) * 64;    // column half within gate
    const int m    = lane & 15;
    const int half = lane >> 4;

    // --- init state from h0/c0 ---
    {
        const int row = tid >> 4, j0 = (tid & 15) * 8;
        #pragma unroll
        for (int j = 0; j < 8; ++j) {
            hS[row * Hsz + j0 + j] = h0[(r0 + row) * Hsz + j0 + j];
            cS[row * Hsz + j0 + j] = c0[(r0 + row) * Hsz + j0 + j];
        }
    }
    __syncthreads();

    for (int t = 0; t < Ssz; ++t) {
        // ---- build comb = [x_t | bf16(h)] ----
        {
            const int row = tid >> 4;
            const int i0  = (tid & 15) * 4;
            uint2 xv = *(const uint2*)(xbf + ((size_t)t * Bsz + r0 + row) * Isz + i0);
            *(uint2*)&comb[row * CPAD + i0] = xv;
            const int j0 = (tid & 15) * 8;
            #pragma unroll
            for (int j = 0; j < 8; ++j)
                comb[row * CPAD + Isz + j0 + j] = f2bf(hS[row * Hsz + j0 + j]);
            // prefetch next timestep's 192KB W slab (global_prefetch_b8)
            if (t + 1 < Ssz) {
                const char* nw = (const char*)(wall + (size_t)(t + 1) * WSTEP);
                #pragma unroll
                for (int p = 0; p < 6; ++p)
                    __builtin_prefetch(nw + ((size_t)tid * 6 + p) * 128, 0, 0);
            }
        }
        __syncthreads();

        // ---- GEMM [16x192] x [192x512]: 4 N-tiles x 6 K-steps per wave ----
        v8f acc0 = {}, acc1 = {}, acc2 = {}, acc3 = {};
        const unsigned short* Wg = wall + (size_t)t * WSTEP + (size_t)g * KD * Hsz;
        #pragma unroll
        for (int kk = 0; kk < 6; ++kk) {
            const int kb = kk * 32;
            // A fragment (16-bit A 16x32 layout): lane m = row, K split by lane half
            v8bf lo = *(const v8bf*)&comb[m * CPAD + kb + half * 8];
            v8bf hi = *(const v8bf*)&comb[m * CPAD + kb + 16 + half * 8];
            v16bf a = __builtin_shufflevector(lo, hi,
                        0, 1, 2, 3, 4, 5, 6, 7, 8, 9, 10, 11, 12, 13, 14, 15);
            // B fragments (lane m = column N, K split by lane half in same order as A)
            const unsigned short* wcol = Wg + (size_t)kb * Hsz + colHalf + m;
            v16bf b0, b1, b2, b3;
            #pragma unroll
            for (int s = 0; s < 16; ++s) {
                const int k = (s < 8) ? (half * 8 + s) : (16 + half * 8 + (s - 8));
                const unsigned short* wp = wcol + (size_t)k * Hsz;
                b0[s] = __builtin_bit_cast(__bf16, wp[0]);
                b1[s] = __builtin_bit_cast(__bf16, wp[16]);
                b2[s] = __builtin_bit_cast(__bf16, wp[32]);
                b3[s] = __builtin_bit_cast(__bf16, wp[48]);
            }
            acc0 = __builtin_amdgcn_wmma_f32_16x16x32_bf16(false, a, false, b0, (short)0, acc0, false, false);
            acc1 = __builtin_amdgcn_wmma_f32_16x16x32_bf16(false, a, false, b1, (short)0, acc1, false, false);
            acc2 = __builtin_amdgcn_wmma_f32_16x16x32_bf16(false, a, false, b2, (short)0, acc2, false, false);
            acc3 = __builtin_amdgcn_wmma_f32_16x16x32_bf16(false, a, false, b3, (short)0, acc3, false, false);
        }
        // ---- spill D tiles to LDS (C/D layout: M = v + 8*half, N = m) ----
        {
            const int nb = g * Hsz + colHalf + m;
            #pragma unroll
            for (int v = 0; v < 8; ++v) {
                const int M = v + 8 * half;
                gO[M * ND + nb +  0] = acc0[v];
                gO[M * ND + nb + 16] = acc1[v];
                gO[M * ND + nb + 32] = acc2[v];
                gO[M * ND + nb + 48] = acc3[v];
            }
        }
        __syncthreads();

        // ---- gates + state update (trans-op VALU) ----
        {
            const int row = tid >> 4, hbase = (tid & 15) * 8;
            const float* bt = ball + (size_t)t * ND;
            #pragma unroll
            for (int j = 0; j < 8; ++j) {
                const int h = hbase + j;
                float gi = gO[row * ND +   0 + h] + bt[  0 + h];
                float gf = gO[row * ND + 128 + h] + bt[128 + h];
                float gc = gO[row * ND + 256 + h] + bt[256 + h];
                float go = gO[row * ND + 384 + h] + bt[384 + h];
                float it = sigmf(gi), ft = sigmf(gf), ot = sigmf(go);
                float ch = tanhf(gc);
                float cn = fmaf(ft, cS[row * Hsz + h], it * ch);
                cS[row * Hsz + h] = cn;
                hS[row * Hsz + h] = ot * tanhf(cn);
            }
        }
        __syncthreads();
    }

    // ---- emit final h ----
    {
        const int row = tid >> 4, j0 = (tid & 15) * 8;
        #pragma unroll
        for (int j = 0; j < 8; ++j)
            hfin[(r0 + row) * Hsz + j0 + j] = hS[row * Hsz + j0 + j];
    }
}

// ---------------------------------------------------------------------------
// output projection: out = h_T @ (out_w_mu + softplus(out_w_rho)*eps) + b_out
// 256 threads, one batch row each; tiny.
// ---------------------------------------------------------------------------
__global__ __launch_bounds__(256) void k_outproj(const float* __restrict__ hfin,
                                                 const float* __restrict__ ow_mu,
                                                 const float* __restrict__ ow_rho,
                                                 const float* __restrict__ eps_ow,
                                                 const float* __restrict__ ob_mu,
                                                 const float* __restrict__ ob_rho,
                                                 const float* __restrict__ eps_ob,
                                                 float* __restrict__ out) {
    const int b = threadIdx.x;
    float acc[8];
    #pragma unroll
    for (int o = 0; o < 8; ++o) acc[o] = 0.f;
    for (int h = 0; h < Hsz; ++h) {
        const float hv = hfin[b * Hsz + h];
        #pragma unroll
        for (int o = 0; o < 8; ++o) {
            const int wi = h * 8 + o;
            acc[o] = fmaf(hv, fmaf(softplusf(ow_rho[wi]), eps_ow[wi], ow_mu[wi]), acc[o]);
        }
    }
    #pragma unroll
    for (int o = 0; o < 8; ++o)
        out[b * 8 + o] = acc[o] + fmaf(softplusf(ob_rho[o]), eps_ob[o], ob_mu[o]);
}

// ---------------------------------------------------------------------------
extern "C" void kernel_launch(void* const* d_in, const int* in_sizes, int n_in,
                              void* d_out, int out_size, void* d_ws, size_t ws_size,
                              hipStream_t stream) {
    const float* x        = (const float*)d_in[0];
    const float* h0       = (const float*)d_in[1];
    const float* c0       = (const float*)d_in[2];
    const float* w_mu     = (const float*)d_in[3];
    const float* w_rho    = (const float*)d_in[4];
    const float* b_mu     = (const float*)d_in[5];
    const float* b_rho    = (const float*)d_in[6];
    const float* ow_mu    = (const float*)d_in[7];
    const float* ow_rho   = (const float*)d_in[8];
    const float* ob_mu    = (const float*)d_in[9];
    const float* ob_rho   = (const float*)d_in[10];
    const float* eps_w    = (const float*)d_in[11];
    const float* eps_b    = (const float*)d_in[12];
    const float* eps_ow   = (const float*)d_in[13];
    const float* eps_ob   = (const float*)d_in[14];
    float* out = (float*)d_out;

    char* ws = (char*)d_ws;
    unsigned short* wall = (unsigned short*)(ws + OFF_WALL);
    float*          ball = (float*)(ws + OFF_BALL);
    unsigned short* xbf  = (unsigned short*)(ws + OFF_XBF);
    float*          hfin = (float*)(ws + OFF_HFIN);
    float*          wsig = (float*)(ws + OFF_WSIG);

    k_prep_sigma<<<dim3(384),   dim3(256), 0, stream>>>(w_rho, wsig);
    k_prep_w    <<<dim3(98304), dim3(256), 0, stream>>>(w_mu, wsig, eps_w, (unsigned*)wall);
    k_prep_b    <<<dim3(1024),  dim3(256), 0, stream>>>(b_mu, b_rho, eps_b, ball);
    k_prep_x    <<<dim3(16384), dim3(256), 0, stream>>>(x, (unsigned*)xbf);
    k_lstm      <<<dim3(NWG),   dim3(256), 0, stream>>>(xbf, wall, ball, h0, c0, hfin);
    k_outproj   <<<dim3(1),     dim3(256), 0, stream>>>(hfin, ow_mu, ow_rho, eps_ow,
                                                        ob_mu, ob_rho, eps_ob, out);
}
